// VQAttention_5411658793572
// MI455X (gfx1250) — matrix operand order repaired and checked by reading, not compiled, not run on hardware
//
#include <hip/hip_runtime.h>
#include <hip/hip_bf16.h>

// ---------------------------------------------------------------- constants
#define B_  8
#define L_  512
#define M_  512
#define H_  8
#define DK_ 128
#define DV_ 128
#define S_  512
#define DM_ 1024
#define W_  (M_ + L_)           // 1024
#define BH_ (B_ * H_)           // 64
#define INV_TAU 0.08838834764831845f   // 1/sqrt(128)

typedef __attribute__((ext_vector_type(16))) _Float16 v16h;
typedef __attribute__((ext_vector_type(8)))  _Float16 v8h;
typedef __attribute__((ext_vector_type(8)))  float    v8f;
typedef __attribute__((ext_vector_type(4)))  unsigned int v4u;
typedef __attribute__((ext_vector_type(8)))  int      v8i;
typedef __attribute__((ext_vector_type(4)))  int      v4i;

// ---------------------------------------------------------------- TDM helper
// Issue a Tensor Data Mover load of a 2-D f16 tile (rows x row_elems, row
// stride `stride_elems`) from global into LDS at byte address `lds_byte_addr`,
// with hardware LDS padding (pad_interval/pad_amount codes per D# spec).
// D# Group0/Group1 bitfields per CDNA5 ISA ch.8 (CDNA5 ISA).
__device__ __forceinline__ void tdm_load_2d(unsigned lds_byte_addr,
                                            const void* gptr,
                                            int rows, int row_elems,
                                            int stride_elems,
                                            int pad_interval_code,
                                            int pad_amount_code)
{
    const unsigned long long ga = (unsigned long long)gptr;
    v4u g0;
    g0[0] = 1u;                                        // count=1, user descriptor
    g0[1] = lds_byte_addr;                             // lds_addr [63:32]
    g0[2] = (unsigned)(ga & 0xFFFFFFFFu);              // global_addr[31:0]
    g0[3] = (unsigned)((ga >> 32) & 0x01FFFFFFu)       // global_addr[56:32]
          | (2u << 30);                                // type = 2 ("image")
    const unsigned td0 = (unsigned)stride_elems;       // tensor_dim0 (elems)
    const unsigned td1 = 0x40000000u;                  // tensor_dim1: huge
    v8i g1;
    g1[0] = (int)((1u << 16)                           // data_size = 2 bytes
                | (1u << 20)                           // pad_enable
                | ((unsigned)pad_interval_code << 22)
                | ((unsigned)pad_amount_code << 25));
    g1[1] = (int)((td0 & 0xFFFFu) << 16);              // dim0 lo16 @[31:16]
    g1[2] = (int)(((td0 >> 16) & 0xFFFFu)              // dim0 hi16
                | ((td1 & 0xFFFFu) << 16));            // dim1 lo16
    g1[3] = (int)(((td1 >> 16) & 0xFFFFu)              // dim1 hi16
                | ((unsigned)row_elems << 16));        // tile_dim0 @[127:112]
    g1[4] = (int)((unsigned)rows & 0xFFFFu);           // tile_dim1 (tile_dim2=0)
    g1[5] = (int)((unsigned)stride_elems);             // dim0_stride lo32
    g1[6] = 0;                                         // dim0_stride hi / dim1_stride
    g1[7] = 0;
    const v4i z4 = {0, 0, 0, 0};
#if defined(__clang_major__) && (__clang_major__ >= 23)
    const v8i z8 = {0, 0, 0, 0, 0, 0, 0, 0};
    __builtin_amdgcn_tensor_load_to_lds(g0, g1, z4, z4, z8, 0);
#else
    __builtin_amdgcn_tensor_load_to_lds(g0, g1, z4, z4, 0);
#endif
}

// ================================================================ GEMM (WMMA)
// C[batch] = alpha * A[batch] x (transB ? B^T : B) (+ C if accum)
// A: f16 row-major (Mr x K), B: f16 row-major, C: f32 row-major.
// Block tile 64x64, BK=32 (one v_wmma_f32_16x16x32_f16 K-step).
// 128 threads = 4 waves, each wave owns a 32x32 quadrant (2x2 fragments).
// Tile staging: double-buffered TDM tensor_load_to_lds issued by wave 0.
// Per iteration: barrier; issue tile kt+1 into the ping-pong buffer;
// s_wait_tensorcnt<=2 retires tile kt (TENSORcnt completes in order);
// barrier publishes tile kt; WMMA on tile kt overlaps tile kt+1's DMA.
__global__ __launch_bounds__(128)
void gemm_f16_wmma(const _Float16* __restrict__ A, int lda, long long sA,
                   const _Float16* __restrict__ Bm, int ldb, long long sB,
                   int bmod, int transB,
                   float* __restrict__ C, int ldc, long long sC,
                   int K, float alpha, int accum)
{
    __shared__ _Float16 As[2][64][40];     // 64x32 tile, HW pad 8 halfs/row
    __shared__ _Float16 Bs[2][64][40];     // transB path: [n][k], HW pad
    __shared__ _Float16 BsKN[2][32][72];   // !transB path: [k][n], HW pad

    const int tid  = threadIdx.x;
    const int lane = tid & 31;
    const int wave = tid >> 5;
    const int m0 = blockIdx.y * 64;
    const int n0 = blockIdx.x * 64;
    const int batch = blockIdx.z;

    const _Float16* Ab = A  + (long long)batch * sA;
    const _Float16* Bb = Bm + (long long)(batch % bmod) * sB;
    float*          Cb = C  + (long long)batch * sC;

    const unsigned ldsA0 = (unsigned)(unsigned long long)(void*)&As[0][0][0];
    const unsigned ldsA1 = (unsigned)(unsigned long long)(void*)&As[1][0][0];
    const unsigned ldsB0 = (unsigned)(unsigned long long)(void*)&Bs[0][0][0];
    const unsigned ldsB1 = (unsigned)(unsigned long long)(void*)&Bs[1][0][0];
    const unsigned ldsK0 = (unsigned)(unsigned long long)(void*)&BsKN[0][0][0];
    const unsigned ldsK1 = (unsigned)(unsigned long long)(void*)&BsKN[1][0][0];

    v8f acc[2][2];
    for (int i = 0; i < 2; ++i)
        for (int j = 0; j < 2; ++j)
            for (int r = 0; r < 8; ++r) acc[i][j][r] = 0.0f;

    const int wm = (wave >> 1) * 32;
    const int wn = (wave & 1) * 32;
    const int fr = lane & 15;          // row (A) / col (B,C) within fragment
    const int hi = lane >> 4;          // half-wave selector

    const int nk = K >> 5;             // K-steps of 32

    // ---- prologue: issue tile 0 into buffer 0
    if (wave == 0) {
        tdm_load_2d(ldsA0, Ab + (long long)m0 * lda, 64, 32, lda, 3, 3);
        if (transB)
            tdm_load_2d(ldsB0, Bb + (long long)n0 * ldb, 64, 32, ldb, 3, 3);
        else
            tdm_load_2d(ldsK0, Bb + n0, 32, 64, ldb, 4, 3);
    }

    for (int kt = 0; kt < nk; ++kt) {
        const int cur = kt & 1;
        // all waves finished reading buf[cur^1] (iteration kt-1) before we
        // let the TDM overwrite it with tile kt+1
        __syncthreads();
        if (wave == 0) {
            if (kt + 1 < nk) {
                const int k1 = (kt + 1) << 5;
                tdm_load_2d(cur ? ldsA0 : ldsA1,
                            Ab + (long long)m0 * lda + k1, 64, 32, lda, 3, 3);
                if (transB)
                    tdm_load_2d(cur ? ldsB0 : ldsB1,
                                Bb + (long long)n0 * ldb + k1, 64, 32, ldb, 3, 3);
                else
                    tdm_load_2d(cur ? ldsK0 : ldsK1,
                                Bb + (long long)k1 * ldb + n0, 32, 64, ldb, 4, 3);
                __builtin_amdgcn_s_wait_tensorcnt(2);  // tile kt done, kt+1 in flight
            } else {
                __builtin_amdgcn_s_wait_tensorcnt(0);  // last tile done
            }
        }
        __syncthreads();   // publish tile kt to all waves

        // ---- build fragments: row/col = lane&15, contiguous K = 16*hi + t (same order in A and B)
        v16h af[2], bf[2];
        for (int i = 0; i < 2; ++i) {
            const _Float16* ar = &As[cur][wm + i * 16 + fr][0];
            const int kh8 = hi * 16;                // lanes16-31: K 16..31
            for (int t = 0; t < 16; ++t) af[i][t]    = ar[kh8 + t];
        }
        if (transB) {
            for (int j = 0; j < 2; ++j) {
                const _Float16* br = &Bs[cur][wn + j * 16 + fr][0];
                const int ko = hi * 16;             // lanes16-31: K 16..31
                for (int t = 0; t < 16; ++t) bf[j][t] = br[ko + t];
            }
        } else {
            const int ko = hi * 16;
            for (int j = 0; j < 2; ++j) {
                const int nn = wn + j * 16 + fr;
                for (int t = 0; t < 16; ++t) bf[j][t] = BsKN[cur][ko + t][nn];
            }
        }
        for (int i = 0; i < 2; ++i)
            for (int j = 0; j < 2; ++j)
                acc[i][j] = __builtin_amdgcn_wmma_f32_16x16x32_f16(
                    false, af[i], false, bf[j], (short)0, acc[i][j], false, false);
    }

    // ---- store C: VGPR r -> M = r (+8 for lanes 16-31), N = lane&15
    const int mlane = hi * 8;
    for (int i = 0; i < 2; ++i)
        for (int j = 0; j < 2; ++j) {
            const int rbase = m0 + wm + i * 16 + mlane;
            const int cidx  = n0 + wn + j * 16 + fr;
            for (int r = 0; r < 8; ++r) {
                long long off = (long long)(rbase + r) * ldc + cidx;
                float v = alpha * acc[i][j][r];
                if (accum) v += Cb[off];
                Cb[off] = v;
            }
        }
}

// ================================================================ elementwise
__global__ void cast_f32_f16(const float* __restrict__ x, _Float16* __restrict__ y,
                             long long n)
{
    long long i  = (long long)blockIdx.x * blockDim.x + threadIdx.x;
    long long st = (long long)gridDim.x * blockDim.x;
    for (; i < n; i += st) y[i] = (_Float16)x[i];
}

// LayerNorm over `cols`, one block per row, output f16.
__global__ __launch_bounds__(256)
void ln_cast_rows(const float* __restrict__ x, _Float16* __restrict__ y, int cols)
{
    const long long row = blockIdx.x;
    const float* xr = x + row * cols;
    _Float16*   yr = y + row * cols;
    __shared__ float sm[32];
    const int wv = threadIdx.x >> 5, ln = threadIdx.x & 31;
    const int nw = blockDim.x >> 5;

    float s = 0.f;
    for (int c = threadIdx.x; c < cols; c += blockDim.x) s += xr[c];
    for (int o = 16; o; o >>= 1) s += __shfl_xor(s, o, 32);
    if (!ln) sm[wv] = s;
    __syncthreads();
    if (threadIdx.x < 32) {
        float t = (ln < nw) ? sm[ln] : 0.f;
        for (int o = 16; o; o >>= 1) t += __shfl_xor(t, o, 32);
        if (!ln) sm[0] = t;
    }
    __syncthreads();
    const float mu = sm[0] / cols;

    float v = 0.f;
    for (int c = threadIdx.x; c < cols; c += blockDim.x) {
        float d = xr[c] - mu; v += d * d;
    }
    for (int o = 16; o; o >>= 1) v += __shfl_xor(v, o, 32);
    if (!ln) sm[wv] = v;
    __syncthreads();
    if (threadIdx.x < 32) {
        float t = (ln < nw) ? sm[ln] : 0.f;
        for (int o = 16; o; o >>= 1) t += __shfl_xor(t, o, 32);
        if (!ln) sm[1] = t;
    }
    __syncthreads();
    const float rstd = rsqrtf(sm[1] / cols + 1e-6f);
    for (int c = threadIdx.x; c < cols; c += blockDim.x)
        yr[c] = (_Float16)((xr[c] - mu) * rstd);
}

// Per-(b,h,l): LN(q) over DK, add x_u/x_v -> qu/qv f16; LN(k) -> kln f16.
// One wave per row, 8 waves / block.
__global__ __launch_bounds__(256)
void qk_post(const float* __restrict__ qf, const float* __restrict__ kvgf,
             const float* __restrict__ xu, const float* __restrict__ xv,
             _Float16* __restrict__ quh, _Float16* __restrict__ qvh,
             _Float16* __restrict__ klnh)
{
    const int id = blockIdx.x * 8 + (threadIdx.x >> 5);
    const int ln = threadIdx.x & 31;
    const int l = id % L_;
    const int h = (id / L_) % H_;
    const int b = id / (L_ * H_);
    const float* qr = qf   + (long long)(b * L_ + l) * (H_ * DK_) + h * DK_;
    const float* kr = kvgf + (long long)(b * L_ + l) * 3072 + h * DK_;

    float q4[4], k4[4], sq = 0.f, sk = 0.f;
    for (int i = 0; i < 4; ++i) {
        q4[i] = qr[ln + 32 * i]; sq += q4[i];
        k4[i] = kr[ln + 32 * i]; sk += k4[i];
    }
    for (int o = 16; o; o >>= 1) { sq += __shfl_xor(sq, o, 32); sk += __shfl_xor(sk, o, 32); }
    const float muq = sq / DK_, muk = sk / DK_;
    float vq = 0.f, vk = 0.f;
    for (int i = 0; i < 4; ++i) {
        float d = q4[i] - muq; vq += d * d;
        d = k4[i] - muk; vk += d * d;
    }
    for (int o = 16; o; o >>= 1) { vq += __shfl_xor(vq, o, 32); vk += __shfl_xor(vk, o, 32); }
    const float rq = rsqrtf(vq / DK_ + 1e-6f), rk = rsqrtf(vk / DK_ + 1e-6f);
    const long long ro = ((long long)(b * H_ + h) * L_ + l) * DK_;
    for (int i = 0; i < 4; ++i) {
        const int d = ln + 32 * i;
        const float qn = (q4[i] - muq) * rq;
        quh[ro + d]  = (_Float16)(qn + xu[h * DK_ + d]);
        qvh[ro + d]  = (_Float16)(qn + xv[h * DK_ + d]);
        klnh[ro + d] = (_Float16)((k4[i] - muk) * rk);
    }
}

// ||codebook[h,s]||^2, one wave per (h,s)
__global__ __launch_bounds__(256)
void codebook_norm2(const float* __restrict__ cb, float* __restrict__ cn2)
{
    const int id = blockIdx.x * 8 + (threadIdx.x >> 5);
    const int ln = threadIdx.x & 31;
    const float* cr = cb + (long long)id * DK_;
    float s = 0.f;
    for (int i = 0; i < 4; ++i) { float v = cr[ln + 32 * i]; s += v * v; }
    for (int o = 16; o; o >>= 1) s += __shfl_xor(s, o, 32);
    if (!ln) cn2[id] = s;
}

// argmin_s ||k - c_s||^2 == argmax_s (k.c_s - 0.5||c_s||^2); gather codebook
// row into recent_k_hat[bh, M + l, :]. One wave per (b,h,l).
__global__ __launch_bounds__(256)
void vq_select(const float* __restrict__ vqs, const float* __restrict__ cn2,
               const _Float16* __restrict__ cbh, _Float16* __restrict__ rkh)
{
    const int id = blockIdx.x * 8 + (threadIdx.x >> 5);
    const int ln = threadIdx.x & 31;
    const int l  = id % L_;
    const int bh = id / L_;
    const int h  = bh % H_;
    const float* sr = vqs + ((long long)bh * L_ + l) * S_;
    const float* c2 = cn2 + h * S_;
    float best = -3.4e38f; int bi = 0;
    for (int s = ln; s < S_; s += 32) {
        const float v = sr[s] - 0.5f * c2[s];
        if (v > best) { best = v; bi = s; }
    }
    for (int o = 16; o; o >>= 1) {
        const float ob = __shfl_xor(best, o, 32);
        const int   oi = __shfl_xor(bi, o, 32);
        if (ob > best || (ob == best && oi < bi)) { best = ob; bi = oi; }
    }
    const _Float16* crow = cbh + ((long long)h * S_ + bi) * DK_;
    _Float16* dst = rkh + ((long long)bh * W_ + M_ + l) * DK_;
    for (int i = ln; i < DK_; i += 32) dst[i] = crow[i];
}

// (64, M, 128) f32 -> rows [0,M) of (64, W, 128) f16
__global__ void copy_xl_recent(const float* __restrict__ src, _Float16* __restrict__ dst)
{
    const long long n = (long long)BH_ * M_ * DK_;
    long long i  = (long long)blockIdx.x * blockDim.x + threadIdx.x;
    long long st = (long long)gridDim.x * blockDim.x;
    for (; i < n; i += st) {
        const long long bh = i >> 16;           // / (M_*DK_)
        const long long rem = i & 65535;
        dst[bh * (long long)W_ * DK_ + rem] = (_Float16)src[i];
    }
}

// v slice of kvg (b,l,h,d) -> recent_v[bh, M + l, d] f16
__global__ void copy_v_new(const float* __restrict__ kvgf, _Float16* __restrict__ rv)
{
    const long long n = (long long)BH_ * L_ * DV_;
    long long i  = (long long)blockIdx.x * blockDim.x + threadIdx.x;
    long long st = (long long)gridDim.x * blockDim.x;
    for (; i < n; i += st) {
        const int d  = (int)(i & 127);
        const int l  = (int)((i >> 7) & 511);
        const int bh = (int)(i >> 16);
        const int b = bh >> 3, h = bh & 7;
        rv[((long long)bh * W_ + M_ + l) * DV_ + d] =
            (_Float16)kvgf[(long long)(b * L_ + l) * 3072 + 1024 + h * DV_ + d];
    }
}

// Joint masked softmax over recent(1024, rel-shifted bd) + cache(512, log bias).
// One block of 256 per (bh,l) row: exact two-pass max/denominator, f16 probs.
__global__ __launch_bounds__(256)
void softmax_joint(const float* __restrict__ ac, const float* __restrict__ bd,
                   const float* __restrict__ cs, const float* __restrict__ agg_lower,
                   _Float16* __restrict__ pr, _Float16* __restrict__ pc)
{
    const int l  = blockIdx.x % L_;
    const int bh = blockIdx.x / L_;
    const float* acr = ac + ((long long)bh * L_ + l) * W_;
    const float* bdr = bd + ((long long)bh * L_ + l) * W_;
    const float* csr = cs + ((long long)bh * L_ + l) * S_;
    const float* alr = agg_lower + (long long)bh * S_;
    _Float16* prr = pr + ((long long)bh * L_ + l) * W_;
    _Float16* pcr = pc + ((long long)bh * L_ + l) * S_;

    const int t = threadIdx.x;
    float rv[4], cv[2], mx = -1e30f;
    for (int i = 0; i < 4; ++i) {
        const int w = t + 256 * i;
        float v = -1e30f;
        if (w <= l + M_) v = acr[w] + bdr[w + (L_ - 1) - l];   // rel_shift
        rv[i] = v; mx = fmaxf(mx, v);
    }
    for (int i = 0; i < 2; ++i) {
        const int s = t + 256 * i;
        const float lo = alr[s];
        const float bias = (lo > 0.f) ? logf(fmaxf(lo, 1e-30f)) : -1e30f;
        const float v = csr[s] + bias;
        cv[i] = v; mx = fmaxf(mx, v);
    }
    __shared__ float sm[32];
    const int wv = t >> 5, ln = t & 31;
    for (int o = 16; o; o >>= 1) mx = fmaxf(mx, __shfl_xor(mx, o, 32));
    if (!ln) sm[wv] = mx;
    __syncthreads();
    if (t < 32) {
        float v = (ln < 8) ? sm[ln] : -1e30f;
        for (int o = 16; o; o >>= 1) v = fmaxf(v, __shfl_xor(v, o, 32));
        if (!ln) sm[0] = v;
    }
    __syncthreads();
    const float m = sm[0];
    float ssum = 0.f;
    for (int i = 0; i < 4; ++i) { rv[i] = __expf(rv[i] - m); ssum += rv[i]; }
    for (int i = 0; i < 2; ++i) { cv[i] = __expf(cv[i] - m); ssum += cv[i]; }
    __syncthreads();
    for (int o = 16; o; o >>= 1) ssum += __shfl_xor(ssum, o, 32);
    if (!ln) sm[wv] = ssum;
    __syncthreads();
    if (t < 32) {
        float v = (ln < 8) ? sm[ln] : 0.f;
        for (int o = 16; o; o >>= 1) v += __shfl_xor(v, o, 32);
        if (!ln) sm[1] = v;
    }
    __syncthreads();
    const float inv = 1.f / sm[1];
    for (int i = 0; i < 4; ++i) prr[t + 256 * i] = (_Float16)(rv[i] * inv);
    for (int i = 0; i < 2; ++i) pcr[t + 256 * i] = (_Float16)(cv[i] * inv);
}

// wg[b,l,h*DV+d] = wv[bh,l,d] * silu(g[b,l,h*DV+d])  (f16 out for final GEMM)
__global__ void gate_mul(const float* __restrict__ wvb, const float* __restrict__ kvgf,
                         _Float16* __restrict__ wg)
{
    const long long n = (long long)B_ * L_ * H_ * DV_;
    long long i  = (long long)blockIdx.x * blockDim.x + threadIdx.x;
    long long st = (long long)gridDim.x * blockDim.x;
    for (; i < n; i += st) {
        const int d = (int)(i & 127);
        const int h = (int)((i >> 7) & 7);
        const int l = (int)((i >> 10) & 511);
        const int b = (int)(i >> 19);
        const float g = kvgf[(long long)(b * L_ + l) * 3072 + 2048 + h * DV_ + d];
        const float sg = g / (1.f + __expf(-g));
        const float v  = wvb[(((long long)(b * H_ + h)) * L_ + l) * DV_ + d];
        wg[(long long)(b * L_ + l) * (H_ * DV_) + h * DV_ + d] = (_Float16)(v * sg);
    }
}

// ================================================================ launcher
extern "C" void kernel_launch(void* const* d_in, const int* in_sizes, int n_in,
                              void* d_out, int out_size, void* d_ws, size_t ws_size,
                              hipStream_t stream)
{
    (void)in_sizes; (void)n_in; (void)out_size; (void)ws_size;
    const float* x_in      = (const float*)d_in[0];
    const float* xl_k_hat  = (const float*)d_in[2];
    const float* xl_v      = (const float*)d_in[3];
    const float* agg_upper = (const float*)d_in[4];
    const float* agg_lower = (const float*)d_in[5];
    const float* W_q       = (const float*)d_in[6];
    const float* W_kvg     = (const float*)d_in[7];
    const float* W_res     = (const float*)d_in[8];
    const float* x_u       = (const float*)d_in[9];
    const float* x_v       = (const float*)d_in[10];
    const float* xl_r      = (const float*)d_in[11];
    const float* codebook  = (const float*)d_in[12];

    char* wsp = (char*)d_ws;
    auto carve = [&](size_t bytes) -> char* {
        char* p = wsp; wsp += (bytes + 255) & ~(size_t)255; return p;
    };
    const long long R  = (long long)B_ * L_;          // 4096
    _Float16* xh     = (_Float16*)carve(R * DM_ * 2);
    _Float16* wqh    = (_Float16*)carve((long long)DM_ * 1024 * 2);
    _Float16* wkvgh  = (_Float16*)carve((long long)DM_ * 3072 * 2);
    _Float16* wresh  = (_Float16*)carve((long long)1024 * DM_ * 2);
    _Float16* cbh    = (_Float16*)carve((long long)H_ * S_ * DK_ * 2);
    _Float16* xlrh   = (_Float16*)carve((long long)H_ * W_ * DK_ * 2);
    _Float16* agguh  = (_Float16*)carve((long long)BH_ * S_ * DV_ * 2);
    float*    qf32   = (float*)   carve(R * 1024 * 4);
    float*    kvgf   = (float*)   carve(R * 3072 * 4);
    _Float16* quh    = (_Float16*)carve((long long)BH_ * L_ * DK_ * 2);
    _Float16* qvh    = (_Float16*)carve((long long)BH_ * L_ * DK_ * 2);
    _Float16* klnh   = (_Float16*)carve((long long)BH_ * L_ * DK_ * 2);
    float*    vqs    = (float*)   carve((long long)BH_ * L_ * S_ * 4);
    float*    cn2    = (float*)   carve((long long)H_ * S_ * 4);
    _Float16* rkh    = (_Float16*)carve((long long)BH_ * W_ * DK_ * 2);
    _Float16* rvh    = (_Float16*)carve((long long)BH_ * W_ * DV_ * 2);
    float*    ac     = (float*)   carve((long long)BH_ * L_ * W_ * 4);
    float*    bd     = (float*)   carve((long long)BH_ * L_ * W_ * 4);
    float*    cs     = (float*)   carve((long long)BH_ * L_ * S_ * 4);
    _Float16* prh    = (_Float16*)carve((long long)BH_ * L_ * W_ * 2);
    _Float16* pch    = (_Float16*)carve((long long)BH_ * L_ * S_ * 2);
    float*    wvb    = (float*)   carve((long long)BH_ * L_ * DV_ * 4);
    _Float16* wgh    = (_Float16*)carve(R * (long long)(H_ * DV_) * 2);

    const dim3 blk128(128), blk256(256);

    // --- weight / constant casts to f16
    cast_f32_f16<<<4096, blk256, 0, stream>>>(W_q,   wqh,   (long long)DM_ * 1024);
    cast_f32_f16<<<8192, blk256, 0, stream>>>(W_kvg, wkvgh, (long long)DM_ * 3072);
    cast_f32_f16<<<4096, blk256, 0, stream>>>(W_res, wresh, (long long)1024 * DM_);
    cast_f32_f16<<<2048, blk256, 0, stream>>>(codebook, cbh, (long long)H_ * S_ * DK_);
    cast_f32_f16<<<4096, blk256, 0, stream>>>(xl_r, xlrh, (long long)H_ * W_ * DK_);
    cast_f32_f16<<<8192, blk256, 0, stream>>>(agg_upper, agguh, (long long)BH_ * S_ * DV_);

    // --- LN(x) -> f16
    ln_cast_rows<<<(unsigned)R, blk256, 0, stream>>>(x_in, xh, DM_);

    // --- projections: q = xh @ Wq, kvg = xh @ Wkvg
    gemm_f16_wmma<<<dim3(1024 / 64, R / 64, 1), blk128, 0, stream>>>(
        xh, DM_, 0, wqh, 1024, 0, 1, 0, qf32, 1024, 0, DM_, 1.f, 0);
    gemm_f16_wmma<<<dim3(3072 / 64, R / 64, 1), blk128, 0, stream>>>(
        xh, DM_, 0, wkvgh, 3072, 0, 1, 0, kvgf, 3072, 0, DM_, 1.f, 0);

    // --- per-head LN(q), LN(k); add x_u/x_v
    qk_post<<<(BH_ * L_) / 8, blk256, 0, stream>>>(qf32, kvgf, x_u, x_v, quh, qvh, klnh);

    // --- VQ: scores = kln @ cb^T (batched over bh, cb shared per head)
    codebook_norm2<<<(H_ * S_) / 8, blk256, 0, stream>>>(codebook, cn2);
    gemm_f16_wmma<<<dim3(S_ / 64, L_ / 64, BH_), blk128, 0, stream>>>(
        klnh, DK_, (long long)L_ * DK_, cbh, DK_, (long long)S_ * DK_, H_, 1,
        vqs, S_, (long long)L_ * S_, DK_, 1.f, 0);
    vq_select<<<(BH_ * L_) / 8, blk256, 0, stream>>>(vqs, cn2, cbh, rkh);

    // --- recent buffers: [xl_k_hat ; k_hat], [xl_v ; v]
    copy_xl_recent<<<8192, blk256, 0, stream>>>(xl_k_hat, rkh);
    copy_xl_recent<<<8192, blk256, 0, stream>>>(xl_v, rvh);
    copy_v_new<<<8192, blk256, 0, stream>>>(kvgf, rvh);

    // --- scores: ac = qu@rk^T/TAU, bd = qv@xlr^T/TAU, cs = qu@cb^T/TAU
    gemm_f16_wmma<<<dim3(W_ / 64, L_ / 64, BH_), blk128, 0, stream>>>(
        quh, DK_, (long long)L_ * DK_, rkh, DK_, (long long)W_ * DK_, BH_, 1,
        ac, W_, (long long)L_ * W_, DK_, INV_TAU, 0);
    gemm_f16_wmma<<<dim3(W_ / 64, L_ / 64, BH_), blk128, 0, stream>>>(
        qvh, DK_, (long long)L_ * DK_, xlrh, DK_, (long long)W_ * DK_, H_, 1,
        bd, W_, (long long)L_ * W_, DK_, INV_TAU, 0);
    gemm_f16_wmma<<<dim3(S_ / 64, L_ / 64, BH_), blk128, 0, stream>>>(
        quh, DK_, (long long)L_ * DK_, cbh, DK_, (long long)S_ * DK_, H_, 1,
        cs, S_, (long long)L_ * S_, DK_, INV_TAU, 0);

    // --- rel-shift + mask + joint softmax -> f16 probabilities
    softmax_joint<<<BH_ * L_, blk256, 0, stream>>>(ac, bd, cs, agg_lower, prh, pch);

    // --- wv = pr @ recent_v + pc @ agg_upper
    gemm_f16_wmma<<<dim3(DV_ / 64, L_ / 64, BH_), blk128, 0, stream>>>(
        prh, W_, (long long)L_ * W_, rvh, DV_, (long long)W_ * DV_, BH_, 0,
        wvb, DV_, (long long)L_ * DV_, W_, 1.f, 0);
    gemm_f16_wmma<<<dim3(DV_ / 64, L_ / 64, BH_), blk128, 0, stream>>>(
        pch, S_, (long long)L_ * S_, agguh, DV_, (long long)S_ * DV_, BH_, 0,
        wvb, DV_, (long long)L_ * DV_, S_, 1.f, 1);

    // --- gate, output projection into d_out (f32)
    gate_mul<<<8192, blk256, 0, stream>>>(wvb, kvgf, wgh);
    gemm_f16_wmma<<<dim3(DM_ / 64, R / 64, 1), blk128, 0, stream>>>(
        wgh, H_ * DV_, 0, wresh, DM_, 0, 1, 0, (float*)d_out, DM_, 0,
        H_ * DV_, 1.f, 0);
}
